// Qwen2MoEDecoderLayer_22668837388788
// MI455X (gfx1250) — compile-verified, matrix operand order checked by reading, not run on hardware
//
#include <hip/hip_runtime.h>
#include <hip/hip_bf16.h>

#define BB  2
#define SSQ 2048
#define DDM 2048
#define HHN 16
#define DHH 128
#define FFI 5504
#define BSS (BB * SSQ)

typedef __attribute__((ext_vector_type(16))) _Float16 v16h;
typedef __attribute__((ext_vector_type(4)))  _Float16 v4h;
typedef __attribute__((ext_vector_type(8)))  float    v8f;
typedef __attribute__((ext_vector_type(4)))  unsigned int u32x4;
typedef __attribute__((ext_vector_type(8)))  int      i32x8;
typedef __attribute__((ext_vector_type(4)))  int      i32x4;

#if defined(__gfx1250__) && __has_builtin(__builtin_amdgcn_tensor_load_to_lds) && \
    __has_builtin(__builtin_amdgcn_s_wait_tensorcnt)
#define USE_TDM 1
#else
#define USE_TDM 0
#endif

__device__ inline v8f zero8() {
  v8f z = {0.f, 0.f, 0.f, 0.f, 0.f, 0.f, 0.f, 0.f};
  return z;
}

__device__ inline v8f wmma16(v16h a, v16h b, v8f c) {
  return __builtin_amdgcn_wmma_f32_16x16x32_f16(false, a, false, b, (short)0, c,
                                                false, false);
}

// 16-bit A/B fragment (16x32 MxK) from a row-major [rows, ld] f16 source.
// ISA 7.12.2: lanes 0-15 hold row=lane, K=0..7 & 16..23; lanes 16-31 hold
// K=8..15 & 24..31.
__device__ inline v16h frag_h(const _Float16* p, int ld, int rowBase, int kBase,
                              int lane) {
  int ll = lane & 15, sel = (lane >> 4) & 1;
  const _Float16* r = p + (size_t)(rowBase + ll) * (size_t)ld + (kBase + sel * 8);
  v16h f;
#pragma unroll
  for (int i = 0; i < 8; ++i) {
    f[i] = r[i];
    f[i + 8] = r[i + 16];
  }
  return f;
}

// C = A[M,K] x W[N,K]^T, all f16, one wave computes 32x64 as 2x4 WMMA tiles.
__device__ inline void gemm2x4(const _Float16* A, const _Float16* W, int K,
                               int m0, int n0, int lane, v8f c[2][4]) {
  for (int k = 0; k < K; k += 32) {
    __builtin_prefetch(A + (size_t)(m0 + (lane & 15)) * K + k + 128, 0, 1);
    __builtin_prefetch(W + (size_t)(n0 + (lane & 15)) * K + k + 128, 0, 1);
    v16h a0 = frag_h(A, K, m0, k, lane);
    v16h a1 = frag_h(A, K, m0 + 16, k, lane);
    v16h b0 = frag_h(W, K, n0, k, lane);
    v16h b1 = frag_h(W, K, n0 + 16, k, lane);
    v16h b2 = frag_h(W, K, n0 + 32, k, lane);
    v16h b3 = frag_h(W, K, n0 + 48, k, lane);
    c[0][0] = wmma16(a0, b0, c[0][0]);
    c[0][1] = wmma16(a0, b1, c[0][1]);
    c[0][2] = wmma16(a0, b2, c[0][2]);
    c[0][3] = wmma16(a0, b3, c[0][3]);
    c[1][0] = wmma16(a1, b0, c[1][0]);
    c[1][1] = wmma16(a1, b1, c[1][1]);
    c[1][2] = wmma16(a1, b2, c[1][2]);
    c[1][3] = wmma16(a1, b3, c[1][3]);
  }
}

// ---------------- f32 -> f16 weight pre-conversion ----------------
__global__ void __launch_bounds__(256) k_cvt16(const float* __restrict__ in,
                                               _Float16* __restrict__ out,
                                               int n4) {
  int i = blockIdx.x * 256 + threadIdx.x;
  if (i < n4) {
    float4 v = *(const float4*)(in + (size_t)i * 4);
    v4h h = {(_Float16)v.x, (_Float16)v.y, (_Float16)v.z, (_Float16)v.w};
    *(v4h*)(out + (size_t)i * 4) = h;
  }
}

// ---------------- RMSNorm: f32 in -> f16 out ----------------
__global__ void __launch_bounds__(256) k_rmsnorm(const float* __restrict__ x,
                                                 const float* __restrict__ w,
                                                 _Float16* __restrict__ out) {
  int row = blockIdx.x, tid = threadIdx.x;
  const float* xr = x + (size_t)row * DDM;
  float ss = 0.f;
  for (int i = tid; i < DDM; i += 256) {
    float v = xr[i];
    ss += v * v;
  }
  __shared__ float red[256];
  red[tid] = ss;
  __syncthreads();
  for (int s = 128; s > 0; s >>= 1) {
    if (tid < s) red[tid] += red[tid + s];
    __syncthreads();
  }
  float rs = rsqrtf(red[0] * (1.0f / DDM) + 1e-6f);
  for (int i = tid; i < DDM; i += 256)
    out[(size_t)row * DDM + i] = (_Float16)(xr[i] * rs * w[i]);
}

// ---------------- QKV projection: f16 out + bias (2x4 tiles) ----------------
__global__ void __launch_bounds__(256) k_gemm_qkv(const _Float16* __restrict__ A,
                                                  const _Float16* __restrict__ W,
                                                  const float* __restrict__ bias,
                                                  _Float16* __restrict__ out,
                                                  int N, int K) {
  int lane = threadIdx.x & 31, wv = threadIdx.x >> 5;
  int m0 = blockIdx.y * 128 + (wv & 3) * 32;
  int n0 = blockIdx.x * 128 + (wv >> 2) * 64;
  v8f c[2][4];
#pragma unroll
  for (int i = 0; i < 2; ++i)
#pragma unroll
    for (int j = 0; j < 4; ++j) c[i][j] = zero8();
  gemm2x4(A, W, K, m0, n0, lane, c);
  int ll = lane & 15, hf = (lane >> 4) & 1;
#pragma unroll
  for (int i = 0; i < 2; ++i)
#pragma unroll
    for (int j = 0; j < 4; ++j) {
      int col = n0 + j * 16 + ll;
      float bc = bias[col];
#pragma unroll
      for (int jj = 0; jj < 8; ++jj) {
        size_t row = m0 + i * 16 + hf * 8 + jj;
        out[row * N + col] = (_Float16)(c[i][j][jj] + bc);
      }
    }
}

// ---------------- o-proj: out_f32 = residual + A x Wo^T ----------------
__global__ void __launch_bounds__(256) k_gemm_oproj(const _Float16* __restrict__ A,
                                                    const _Float16* __restrict__ W,
                                                    const float* __restrict__ res,
                                                    float* __restrict__ out,
                                                    int N, int K) {
  int lane = threadIdx.x & 31, wv = threadIdx.x >> 5;
  int m0 = blockIdx.y * 128 + (wv & 3) * 32;
  int n0 = blockIdx.x * 128 + (wv >> 2) * 64;
  v8f c[2][4];
#pragma unroll
  for (int i = 0; i < 2; ++i)
#pragma unroll
    for (int j = 0; j < 4; ++j) c[i][j] = zero8();
  gemm2x4(A, W, K, m0, n0, lane, c);
  int ll = lane & 15, hf = (lane >> 4) & 1;
#pragma unroll
  for (int i = 0; i < 2; ++i)
#pragma unroll
    for (int j = 0; j < 4; ++j) {
      int col = n0 + j * 16 + ll;
#pragma unroll
      for (int jj = 0; jj < 8; ++jj) {
        size_t row = m0 + i * 16 + hf * 8 + jj;
        out[row * N + col] = res[row * N + col] + c[i][j][jj];
      }
    }
}

// -------- MoE down-proj: out = base + mask[row] * (t x Wd^T) (2x4 tiles) --------
__global__ void __launch_bounds__(256) k_gemm_down(const _Float16* __restrict__ A,
                                                   const _Float16* __restrict__ W,
                                                   const float* __restrict__ base,
                                                   const float* __restrict__ mask,
                                                   float* __restrict__ out,
                                                   int N, int K) {
  int lane = threadIdx.x & 31, wv = threadIdx.x >> 5;
  int m0 = blockIdx.y * 128 + (wv & 3) * 32;
  int n0 = blockIdx.x * 128 + (wv >> 2) * 64;
  v8f c[2][4];
#pragma unroll
  for (int i = 0; i < 2; ++i)
#pragma unroll
    for (int j = 0; j < 4; ++j) c[i][j] = zero8();
  gemm2x4(A, W, K, m0, n0, lane, c);
  int ll = lane & 15, hf = (lane >> 4) & 1;
#pragma unroll
  for (int i = 0; i < 2; ++i)
#pragma unroll
    for (int j = 0; j < 4; ++j) {
      int col = n0 + j * 16 + ll;
#pragma unroll
      for (int jj = 0; jj < 8; ++jj) {
        size_t row = m0 + i * 16 + hf * 8 + jj;
        out[row * N + col] = base[row * N + col] + mask[row] * c[i][j][jj];
      }
    }
}

// ------- fused gate/up: t = silu(x Wg^T) * (x Wu^T), f16 out (2x2 dual) -------
__global__ void __launch_bounds__(256) k_gemm_gateup(const _Float16* __restrict__ A,
                                                     const _Float16* __restrict__ Wg,
                                                     const _Float16* __restrict__ Wu,
                                                     _Float16* __restrict__ out,
                                                     int N, int K) {
  int lane = threadIdx.x & 31, wv = threadIdx.x >> 5;
  int m0 = blockIdx.y * 128 + (wv & 3) * 32;
  int n0 = blockIdx.x * 64 + (wv >> 2) * 32;
  v8f cg[2][2], cu[2][2];
#pragma unroll
  for (int i = 0; i < 2; ++i)
#pragma unroll
    for (int j = 0; j < 2; ++j) {
      cg[i][j] = zero8();
      cu[i][j] = zero8();
    }
  for (int k = 0; k < K; k += 32) {
    v16h a0 = frag_h(A, K, m0, k, lane);
    v16h a1 = frag_h(A, K, m0 + 16, k, lane);
    v16h g0 = frag_h(Wg, K, n0, k, lane);
    v16h g1 = frag_h(Wg, K, n0 + 16, k, lane);
    v16h u0 = frag_h(Wu, K, n0, k, lane);
    v16h u1 = frag_h(Wu, K, n0 + 16, k, lane);
    cg[0][0] = wmma16(a0, g0, cg[0][0]);
    cg[0][1] = wmma16(a0, g1, cg[0][1]);
    cg[1][0] = wmma16(a1, g0, cg[1][0]);
    cg[1][1] = wmma16(a1, g1, cg[1][1]);
    cu[0][0] = wmma16(a0, u0, cu[0][0]);
    cu[0][1] = wmma16(a0, u1, cu[0][1]);
    cu[1][0] = wmma16(a1, u0, cu[1][0]);
    cu[1][1] = wmma16(a1, u1, cu[1][1]);
  }
  int ll = lane & 15, hf = (lane >> 4) & 1;
#pragma unroll
  for (int i = 0; i < 2; ++i)
#pragma unroll
    for (int j = 0; j < 2; ++j) {
      int col = n0 + j * 16 + ll;
#pragma unroll
      for (int jj = 0; jj < 8; ++jj) {
        size_t row = m0 + i * 16 + hf * 8 + jj;
        float g = cg[i][j][jj];
        float u = cu[i][j][jj];
        out[row * N + col] = (_Float16)(g * (1.0f / (1.0f + __expf(-g))) * u);
      }
    }
}

// ---------------- RoPE (in place on f16 Q,K) ----------------
__global__ void __launch_bounds__(256) k_rope(_Float16* __restrict__ q,
                                              _Float16* __restrict__ kk) {
  int idx = blockIdx.x * 256 + threadIdx.x;
  if (idx >= BSS * HHN * (DHH / 2)) return;
  int i = idx & 63;
  int h = (idx >> 6) & (HHN - 1);
  int tok = idx >> 10;
  int s = tok & (SSQ - 1);
  float freq = (float)s * __expf(-((float)(2 * i) / (float)DHH) * 9.210340371976184f);
  float cs = __cosf(freq), sn = __sinf(freq);
  size_t base = (size_t)tok * (HHN * DHH) + (size_t)h * DHH;
  float a = (float)q[base + i], b = (float)q[base + i + 64];
  q[base + i] = (_Float16)(a * cs - b * sn);
  q[base + i + 64] = (_Float16)(b * cs + a * sn);
  float ka = (float)kk[base + i], kb = (float)kk[base + i + 64];
  kk[base + i] = (_Float16)(ka * cs - kb * sn);
  kk[base + i + 64] = (_Float16)(kb * cs + ka * sn);
}

#if USE_TDM
// TDM: DMA a 32x128-half tile (rows stride 2048 halfs) from global into LDS.
// D# per ISA 8.3/8.4: group0 = {count, lds_addr, global_addr, type=2},
// group1 = {data_size=2B, tensor dims/strides, tile 128x32}.
__device__ inline void tdm_load_tile_32x128h(const _Float16* gsrc,
                                             unsigned lds_byte_off) {
  unsigned long long ga = (unsigned long long)(uintptr_t)gsrc;
  u32x4 g0;
  g0[0] = 1u;                                   // count=1, user descriptor
  g0[1] = lds_byte_off;                         // lds_addr (bytes)
  g0[2] = (unsigned)(ga & 0xffffffffu);         // global_addr[31:0]
  g0[3] = (unsigned)((ga >> 32) & 0x01ffffffu)  // global_addr[56:32]
          | (2u << 30);                         // type=2 ("image")
  i32x8 g1;
  g1[0] = 0x00010000;                           // data_size=1 (2 bytes)
  g1[1] = (int)((2048u & 0xffffu) << 16);       // tensor_dim0 lo16
  g1[2] = (int)((2048u & 0xffffu) << 16);       // dim0 hi16=0 | tensor_dim1 lo16
  g1[3] = (int)(128u << 16);                    // dim1 hi16=0 | tile_dim0=128
  g1[4] = 32;                                   // tile_dim1=32, tile_dim2=0
  g1[5] = 2048;                                 // tensor_dim0_stride lo32
  g1[6] = 0;                                    // stride hi | dim1_stride lo
  g1[7] = 0;
  i32x4 z4 = {0, 0, 0, 0};
#if __clang_major__ >= 23
  i32x8 z8 = {0, 0, 0, 0, 0, 0, 0, 0};
  __builtin_amdgcn_tensor_load_to_lds(g0, g1, z4, z4, z8, 0);
#else
  __builtin_amdgcn_tensor_load_to_lds(g0, g1, z4, z4, 0);
#endif
}
#endif

// ---------------- flash attention, causal, per (b,h, 128-row q block) ----------------
__global__ void __launch_bounds__(256) k_attn(const _Float16* __restrict__ q,
                                              const _Float16* __restrict__ k,
                                              const _Float16* __restrict__ v,
                                              _Float16* __restrict__ ctx) {
  __shared__ __align__(16) _Float16 kbuf[32 * 128];   // [key][dh]
  __shared__ __align__(16) _Float16 vbuf[128 * 32];   // [dh][key] (transposed)
  __shared__ __align__(16) _Float16 pbuf[8][16 * 32]; // per-wave P tile
  int tid = threadIdx.x, lane = tid & 31, wv = tid >> 5;
  int qb = blockIdx.x;
  int bh = blockIdx.y;
  int b = bh >> 4, h = bh & (HHN - 1);
  const _Float16* qbase = q + (size_t)b * SSQ * (HHN * DHH) + (size_t)h * DHH;
  const _Float16* kbase = k + (size_t)b * SSQ * (HHN * DHH) + (size_t)h * DHH;
  const _Float16* vbase = v + (size_t)b * SSQ * (HHN * DHH) + (size_t)h * DHH;
  int q0 = qb * 128 + wv * 16;
  int ll = lane & 15, hf = (lane >> 4) & 1;

  v8f o[8];
#pragma unroll
  for (int t = 0; t < 8; ++t) o[t] = zero8();
  float rm[8], rl[8];
#pragma unroll
  for (int j = 0; j < 8; ++j) {
    rm[j] = -1e30f;
    rl[j] = 0.f;
  }

  int ntiles = qb * 4 + 4;
  for (int t = 0; t < ntiles; ++t) {
    int k0 = t * 32;
#if USE_TDM
    // K tile via Tensor Data Mover (one wave issues the whole-tile DMA;
    // generic shared address truncated to 32 bits == LDS byte offset, ISA 10.2)
    if (wv == 0) {
      tdm_load_tile_32x128h(kbase + (size_t)k0 * (HHN * DHH),
                            (unsigned)(uintptr_t)&kbuf[0]);
      __builtin_amdgcn_s_wait_tensorcnt(0);
    }
#else
#pragma unroll
    for (int r = 0; r < 2; ++r) {
      int u = tid + r * 256;
      int key = u >> 3, off = (u & 7) * 8;
      *(float4*)&kbuf[key * 128 + off] =
          *(const float4*)&kbase[(size_t)(k0 + key) * (HHN * DHH) + off];
    }
#endif
    // stage V transposed: vbuf[d][key]
    {
      int key = tid & 31;
      int d0 = (tid >> 5) * 16;
      const _Float16* vg = vbase + (size_t)(k0 + key) * (HHN * DHH);
#pragma unroll
      for (int i = 0; i < 16; ++i) vbuf[(d0 + i) * 32 + key] = vg[d0 + i];
    }
    __syncthreads();

    if (k0 <= q0 + 15) {
      v8f s0 = zero8(), s1 = zero8();
      for (int d = 0; d < DHH; d += 32) {
        v16h aq = frag_h(qbase, HHN * DHH, q0, d, lane);
        v16h b0 = frag_h(kbuf, 128, 0, d, lane);
        v16h b1 = frag_h(kbuf, 128, 16, d, lane);
        s0 = wmma16(aq, b0, s0);
        s1 = wmma16(aq, b1, s1);
      }
      float alpha[8];
#pragma unroll
      for (int j = 0; j < 8; ++j) {
        int row = q0 + hf * 8 + j;
        float x0 = s0[j] * 0.08838834764831845f;
        float x1 = s1[j] * 0.08838834764831845f;
        if (k0 + ll > row) x0 = -1e30f;
        if (k0 + 16 + ll > row) x1 = -1e30f;
        float mx = fmaxf(x0, x1);
        for (int md = 1; md < 16; md <<= 1) mx = fmaxf(mx, __shfl_xor(mx, md));
        float nm = fmaxf(rm[j], mx);
        float al = __expf(rm[j] - nm);
        float p0 = __expf(x0 - nm), p1 = __expf(x1 - nm);
        float sum = p0 + p1;
        for (int md = 1; md < 16; md <<= 1) sum += __shfl_xor(sum, md);
        rl[j] = rl[j] * al + sum;
        rm[j] = nm;
        alpha[j] = al;
        pbuf[wv][(hf * 8 + j) * 32 + ll] = (_Float16)p0;
        pbuf[wv][(hf * 8 + j) * 32 + 16 + ll] = (_Float16)p1;
      }
#pragma unroll
      for (int tt = 0; tt < 8; ++tt)
#pragma unroll
        for (int j = 0; j < 8; ++j) o[tt][j] *= alpha[j];
      v16h ap = frag_h(&pbuf[wv][0], 32, 0, 0, lane);
#pragma unroll
      for (int tt = 0; tt < 8; ++tt) {
        v16h bv = frag_h(vbuf, 32, tt * 16, 0, lane);
        o[tt] = wmma16(ap, bv, o[tt]);
      }
    }
    __syncthreads();
  }

#pragma unroll
  for (int tt = 0; tt < 8; ++tt)
#pragma unroll
    for (int j = 0; j < 8; ++j) {
      int row = q0 + hf * 8 + j;
      float val = o[tt][j] / rl[j];
      ctx[((size_t)b * SSQ + row) * (HHN * DHH) + (size_t)h * DHH + tt * 16 + ll] =
          (_Float16)val;
    }
}

// ---------------- router: top-1 over 2 experts -> masks ----------------
__global__ void __launch_bounds__(256) k_router(const _Float16* __restrict__ hn,
                                                const float* __restrict__ rw,
                                                const float* __restrict__ rb,
                                                float* __restrict__ m1,
                                                float* __restrict__ m2) {
  int row = blockIdx.x, tid = threadIdx.x;
  const _Float16* xr = hn + (size_t)row * DDM;
  float d0 = 0.f, d1 = 0.f;
  for (int i = tid; i < DDM; i += 256) {
    float xv = (float)xr[i];
    d0 += xv * rw[i];
    d1 += xv * rw[DDM + i];
  }
  __shared__ float r0[256], r1[256];
  r0[tid] = d0;
  r1[tid] = d1;
  __syncthreads();
  for (int s = 128; s > 0; s >>= 1) {
    if (tid < s) {
      r0[tid] += r0[tid + s];
      r1[tid] += r1[tid + s];
    }
    __syncthreads();
  }
  if (tid == 0) {
    float l0 = r0[0] + rb[0], l1 = r1[0] + rb[1];
    float w1 = (l0 >= l1) ? 1.f : 0.f;
    m1[row] = w1;
    m2[row] = 1.f - w1;
  }
}

extern "C" void kernel_launch(void* const* d_in, const int* in_sizes, int n_in,
                              void* d_out, int out_size, void* d_ws, size_t ws_size,
                              hipStream_t stream) {
  const float* hidden = (const float*)d_in[0];
  const float* ln1w = (const float*)d_in[1];
  const float* wq = (const float*)d_in[2];
  const float* bq = (const float*)d_in[3];
  const float* wk = (const float*)d_in[4];
  const float* bk = (const float*)d_in[5];
  const float* wvv = (const float*)d_in[6];
  const float* bv = (const float*)d_in[7];
  const float* wo = (const float*)d_in[8];
  const float* ln2w = (const float*)d_in[9];
  const float* e1g = (const float*)d_in[10];
  const float* e1u = (const float*)d_in[11];
  const float* e1d = (const float*)d_in[12];
  const float* e2g = (const float*)d_in[13];
  const float* e2u = (const float*)d_in[14];
  const float* e2d = (const float*)d_in[15];
  const float* rw = (const float*)d_in[16];
  const float* rb = (const float*)d_in[17];
  float* out = (float*)d_out;

  char* ws = (char*)d_ws;
  size_t off = 0;
  _Float16* hn16 = (_Float16*)(ws + off); off += (size_t)BSS * DDM * 2;
  _Float16* q16  = (_Float16*)(ws + off); off += (size_t)BSS * DDM * 2;
  _Float16* k16  = (_Float16*)(ws + off); off += (size_t)BSS * DDM * 2;
  _Float16* v16  = (_Float16*)(ws + off); off += (size_t)BSS * DDM * 2;
  _Float16* ctx16 = (_Float16*)(ws + off); off += (size_t)BSS * DDM * 2;
  float* hmid = (float*)(ws + off); off += (size_t)BSS * DDM * 4;
  _Float16* hn2 = (_Float16*)(ws + off); off += (size_t)BSS * DDM * 2;
  _Float16* t16 = (_Float16*)(ws + off); off += (size_t)BSS * FFI * 2;
  _Float16* attW = (_Float16*)(ws + off); off += (size_t)DDM * DDM * 2;   // reused wq/wk/wv/wo
  _Float16* expA = (_Float16*)(ws + off); off += (size_t)FFI * DDM * 2;   // reused gate/down
  _Float16* expB = (_Float16*)(ws + off); off += (size_t)FFI * DDM * 2;   // reused up
  float* mask1 = (float*)(ws + off); off += (size_t)BSS * 4;
  float* mask2 = (float*)(ws + off); off += (size_t)BSS * 4;

  dim3 blk(256);
  dim3 gD(DDM / 128, BSS / 128);   // 2x4-tiled N=2048 GEMMs
  dim3 gGU(FFI / 64, BSS / 128);   // gate/up dual GEMM
  int nAtt4 = (DDM * DDM) / 4;     // f32->f16 convert sizes (in float4 units)
  int nExp4 = (FFI * DDM) / 4;

  // 1) rmsnorm -> hn16
  k_rmsnorm<<<BSS, blk, 0, stream>>>(hidden, ln1w, hn16);
  // 2) QKV projections (weights converted to f16 once each, buffer reused)
  k_cvt16<<<(nAtt4 + 255) / 256, blk, 0, stream>>>(wq, attW, nAtt4);
  k_gemm_qkv<<<gD, blk, 0, stream>>>(hn16, attW, bq, q16, DDM, DDM);
  k_cvt16<<<(nAtt4 + 255) / 256, blk, 0, stream>>>(wk, attW, nAtt4);
  k_gemm_qkv<<<gD, blk, 0, stream>>>(hn16, attW, bk, k16, DDM, DDM);
  k_cvt16<<<(nAtt4 + 255) / 256, blk, 0, stream>>>(wvv, attW, nAtt4);
  k_gemm_qkv<<<gD, blk, 0, stream>>>(hn16, attW, bv, v16, DDM, DDM);
  // 3) RoPE on q,k
  k_rope<<<(BSS * HHN * (DHH / 2)) / 256, blk, 0, stream>>>(q16, k16);
  // 4) causal flash attention -> ctx16
  k_attn<<<dim3(SSQ / 128, BB * HHN), blk, 0, stream>>>(q16, k16, v16, ctx16);
  // 5) o-proj + residual -> hmid (f32)
  k_cvt16<<<(nAtt4 + 255) / 256, blk, 0, stream>>>(wo, attW, nAtt4);
  k_gemm_oproj<<<gD, blk, 0, stream>>>(ctx16, attW, hidden, hmid, DDM, DDM);
  // 6) rmsnorm2 -> hn2
  k_rmsnorm<<<BSS, blk, 0, stream>>>(hmid, ln2w, hn2);
  // 7) router masks
  k_router<<<BSS, blk, 0, stream>>>(hn2, rw, rb, mask1, mask2);
  // 8) expert 1
  k_cvt16<<<(nExp4 + 255) / 256, blk, 0, stream>>>(e1g, expA, nExp4);
  k_cvt16<<<(nExp4 + 255) / 256, blk, 0, stream>>>(e1u, expB, nExp4);
  k_gemm_gateup<<<gGU, blk, 0, stream>>>(hn2, expA, expB, t16, FFI, DDM);
  k_cvt16<<<(nExp4 + 255) / 256, blk, 0, stream>>>(e1d, expA, nExp4);
  k_gemm_down<<<gD, blk, 0, stream>>>(t16, expA, hmid, mask1, out, DDM, FFI);
  // 9) expert 2 (reuses buffers; stream order serializes)
  k_cvt16<<<(nExp4 + 255) / 256, blk, 0, stream>>>(e2g, expA, nExp4);
  k_cvt16<<<(nExp4 + 255) / 256, blk, 0, stream>>>(e2u, expB, nExp4);
  k_gemm_gateup<<<gGU, blk, 0, stream>>>(hn2, expA, expB, t16, FFI, DDM);
  k_cvt16<<<(nExp4 + 255) / 256, blk, 0, stream>>>(e2d, expA, nExp4);
  k_gemm_down<<<gD, blk, 0, stream>>>(t16, expA, out, mask2, out, DDM, FFI);
}
